// MultiLayerMinGRU_56959856280319
// MI455X (gfx1250) — matrix operand order repaired and checked by reading, not compiled, not run on hardware
//
#include <hip/hip_runtime.h>

typedef __bf16 bf16;
typedef __attribute__((ext_vector_type(16))) __bf16 v16bf;
typedef __attribute__((ext_vector_type(8)))  __bf16 v8bf;
typedef __attribute__((ext_vector_type(8)))  float  v8f;

union Frag16 { v16bf v; v8bf h[2]; };

__device__ __forceinline__ float softplus_f(float x) {
    // log(1 + e^x), numerically stable
    return fmaxf(x, 0.f) + log1pf(__expf(-fabsf(x)));
}

// ---------------------------------------------------------------------------
// fp32 -> bf16 conversion (grid-stride)
// ---------------------------------------------------------------------------
__global__ void to_bf16_kernel(const float* __restrict__ src,
                               bf16* __restrict__ dst, size_t n) {
    size_t i = (size_t)blockIdx.x * blockDim.x + threadIdx.x;
    size_t stride = (size_t)gridDim.x * blockDim.x;
    for (; i < n; i += stride) dst[i] = (bf16)src[i];
}

// ---------------------------------------------------------------------------
// C[M,N] = A[M,K](bf16) @ W[K,N](bf16) + bias[N] (+ res[M,N] if HAS_RES)
// block: 256 threads = 8 waves; block tile 128x128; wave tile 32x64
// (2x4 WMMA tiles of 16x16, K-step 32 via v_wmma_f32_16x16x32_bf16)
// ---------------------------------------------------------------------------
template <bool HAS_RES>
__launch_bounds__(256)
__global__ void gemm_bf16_wmma_kernel(const bf16* __restrict__ A,
                                      const bf16* __restrict__ W,
                                      const float* __restrict__ bias,
                                      const float* res,   // may alias out
                                      float* out,
                                      int N, int K) {
    __shared__ bf16 ldsA[128 * 32];   // [m][k], row stride 32
    __shared__ bf16 ldsB[128 * 32];   // transposed: [n][k], row stride 32

    const int tid   = threadIdx.x;
    const int mBase = blockIdx.y * 128;
    const int nBase = blockIdx.x * 128;
    const int lane  = tid & 31;
    const int wave  = tid >> 5;
    const int wm    = wave >> 1;      // 0..3  (M direction, 32 rows each)
    const int wn    = wave & 1;       // 0..1  (N direction, 64 cols each)
    const int laneM = lane & 15;
    const int khalf = lane >> 4;

    v8f acc[2][4];
    #pragma unroll
    for (int a = 0; a < 2; ++a)
        #pragma unroll
        for (int b = 0; b < 4; ++b)
            #pragma unroll
            for (int j = 0; j < 8; ++j) acc[a][b][j] = 0.f;

    // A tile: 128 rows x 32 k; each thread: 16 bf16 (two uint4)
    const int arow  = tid >> 1;          // 0..127
    const int akoff = (tid & 1) * 16;    // 0 or 16
    // B tile: 32 k-rows x 128 n; each thread: 16 bf16 (two uint4), transposed
    const int bkr = tid >> 3;            // 0..31
    const int bnc = (tid & 7) * 16;      // 0..112

    for (int k0 = 0; k0 < K; k0 += 32) {
        const uint4* ag = (const uint4*)(A + (size_t)(mBase + arow) * K + k0 + akoff);
        uint4 a0 = ag[0];
        uint4 a1 = ag[1];
        const uint4* wg = (const uint4*)(W + (size_t)(k0 + bkr) * N + nBase + bnc);
        uint4 w0 = wg[0];
        uint4 w1 = wg[1];
        if (k0 + 32 < K) {
            __builtin_prefetch(A + (size_t)(mBase + arow) * K + k0 + 32, 0, 3);
            __builtin_prefetch(W + (size_t)(k0 + 32 + bkr) * N + nBase + bnc, 0, 3);
        }
        __syncthreads();   // previous iteration's LDS reads complete
        *(uint4*)(ldsA + arow * 32 + akoff)     = a0;
        *(uint4*)(ldsA + arow * 32 + akoff + 8) = a1;
        const bf16* wb0 = (const bf16*)&w0;
        const bf16* wb1 = (const bf16*)&w1;
        #pragma unroll
        for (int j = 0; j < 8; ++j) {
            ldsB[(bnc + j) * 32 + bkr]     = wb0[j];
            ldsB[(bnc + 8 + j) * 32 + bkr] = wb1[j];
        }
        __syncthreads();

        // A frag (16x32 bf16): elems 0..7 -> K = khalf*8 + i,
        //                      elems 8..15 -> K = 16 + khalf*8 + i
        Frag16 af[2], bfr[4];
        #pragma unroll
        for (int im = 0; im < 2; ++im) {
            int m = wm * 32 + im * 16 + laneM;
            af[im].h[0] = *(const v8bf*)(ldsA + m * 32 + khalf * 8);
            af[im].h[1] = *(const v8bf*)(ldsA + m * 32 + 16 + khalf * 8);
        }
        // B frag (32x16 bf16): elems 0..7 -> K = khalf*8 + i, elems 8..15 -> K = 16 + khalf*8 + i
        #pragma unroll
        for (int in = 0; in < 4; ++in) {
            int n = wn * 64 + in * 16 + laneM;
            bfr[in].h[0] = *(const v8bf*)(ldsB + n * 32 + khalf * 8);
            bfr[in].h[1] = *(const v8bf*)(ldsB + n * 32 + 16 + khalf * 8);
        }
        #pragma unroll
        for (int im = 0; im < 2; ++im)
            #pragma unroll
            for (int in = 0; in < 4; ++in)
                acc[im][in] = __builtin_amdgcn_wmma_f32_16x16x32_bf16(
                    false, af[im].v, false, bfr[in].v,
                    (short)0, acc[im][in], false, false);
    }

    // epilogue: D layout — VGPR r: M = r + khalf*8, N = laneM
    #pragma unroll
    for (int im = 0; im < 2; ++im) {
        #pragma unroll
        for (int in = 0; in < 4; ++in) {
            int nG = nBase + wn * 64 + in * 16 + laneM;
            float bv = bias[nG];
            #pragma unroll
            for (int r = 0; r < 8; ++r) {
                int mG = mBase + wm * 32 + im * 16 + r + khalf * 8;
                size_t idx = (size_t)mG * N + nG;
                float v = acc[im][in][r] + bv;
                if (HAS_RES) v += res[idx];
                out[idx] = v;
            }
        }
    }
}

// ---------------------------------------------------------------------------
// rmsnorm over rows of width 2048.
// mode 0: out(bf16) = rmsnorm(x)            (feeds next GEMM)
// mode 1: out(f32)  = rmsnorm(x) + x        (in-place allowed)
// ---------------------------------------------------------------------------
__launch_bounds__(256)
__global__ void rmsnorm_kernel(const float* in, const float* __restrict__ gamma,
                               void* out, int width, int mode) {
    __shared__ float red[256];
    const int row = blockIdx.x;
    const int tid = threadIdx.x;
    const float* rp = in + (size_t)row * width;
    float vals[8];
    float ss = 0.f;
    #pragma unroll
    for (int j = 0; j < 8; ++j) {
        float v = rp[tid + j * 256];
        vals[j] = v;
        ss += v * v;
    }
    red[tid] = ss;
    __syncthreads();
    for (int s = 128; s > 0; s >>= 1) {
        if (tid < s) red[tid] += red[tid + s];
        __syncthreads();
    }
    float norm  = sqrtf(red[0]);
    float scale = sqrtf((float)width) / fmaxf(norm, 1e-12f);
    if (mode == 0) {
        bf16* op = (bf16*)out + (size_t)row * width;
        #pragma unroll
        for (int j = 0; j < 8; ++j) {
            int c = tid + j * 256;
            op[c] = (bf16)(vals[j] * scale * (gamma[c] + 1.f));
        }
    } else {
        float* op = (float*)out + (size_t)row * width;
        #pragma unroll
        for (int j = 0; j < 8; ++j) {
            int c = tid + j * 256;
            op[c] = vals[j] * scale * (gamma[c] + 1.f) + vals[j];
        }
    }
}

// ---------------------------------------------------------------------------
// Heinsen log-space scan. One thread per (batch, channel).
// hg: [B, T, 2*DI] fp32 (hid | gate). out: [B, T, DI] bf16 = exp(log_h)[1:].
// ---------------------------------------------------------------------------
__launch_bounds__(256)
__global__ void scan_kernel(const float* __restrict__ hg,
                            bf16* __restrict__ outbf,
                            int T, int DI2, int DI, int nch) {
    int idx = blockIdx.x * blockDim.x + threadIdx.x;
    if (idx >= nch) return;
    int b = idx / DI;
    int c = idx % DI;
    const float* base = hg + (size_t)b * T * DI2;
    bf16* obase = outbf + (size_t)b * T * DI;
    // padded t=-1 step: a_star=0, value term 0 -> lse state (mx=0, sm=1)
    float astar = 0.f, mx = 0.f, sm = 1.f;
    for (int t = 0; t < T; ++t) {
        float hid  = base[(size_t)t * DI2 + c];
        float gate = base[(size_t)t * DI2 + DI + c];
        float lc = -softplus_f(gate);                          // log_coeff
        float lg = (hid >= 0.f) ? __logf(hid + 0.5f) : -softplus_f(-hid);
        float lv = -softplus_f(-gate) + lg;                    // log_value
        astar += lc;
        float v = lv - astar;
        if (v > mx) { sm = sm * __expf(mx - v) + 1.f; mx = v; }
        else        { sm += __expf(v - mx); }
        float lh = astar + mx + __logf(sm);
        obase[(size_t)t * DI + c] = (bf16)__expf(lh);
    }
}

// ---------------------------------------------------------------------------
extern "C" void kernel_launch(void* const* d_in, const int* in_sizes, int n_in,
                              void* d_out, int out_size, void* d_ws, size_t ws_size,
                              hipStream_t stream) {
    (void)in_sizes; (void)n_in; (void)out_size; (void)ws_size;
    constexpr int Bc = 4, T = 2048, DIMc = 1024, HID = 2048, Lc = 4, DI = 3072;
    constexpr int M = Bc * T;   // 8192 rows

    const float* x   = (const float*)d_in[0];
    const float* Wp  = (const float*)d_in[1];
    const float* bp  = (const float*)d_in[2];
    const float* g1  = (const float*)d_in[3];
    const float* Whg = (const float*)d_in[4];
    const float* bhg = (const float*)d_in[5];
    const float* Wo  = (const float*)d_in[6];
    const float* bo  = (const float*)d_in[7];
    const float* g2  = (const float*)d_in[8];
    const float* gf  = (const float*)d_in[9];
    const float* Wf  = (const float*)d_in[10];
    const float* bfv = (const float*)d_in[11];

    char* ws = (char*)d_ws;
    size_t off = 0;
    auto alloc = [&](size_t bytes) -> char* {
        char* p = ws + off;
        off += (bytes + 255) & ~(size_t)255;
        return p;
    };
    float* h    = (float*)alloc((size_t)M * HID * 4);          // 64 MB
    float* hg   = (float*)alloc((size_t)M * 2 * DI * 4);       // 192 MB
    bf16*  ab   = (bf16*) alloc((size_t)M * DI * 2);           // 48 MB (GEMM A operand)
    bf16*  wp_b = (bf16*) alloc((size_t)DIMc * HID * 2);
    bf16*  whg_b= (bf16*) alloc((size_t)Lc * HID * 2 * DI * 2);
    bf16*  wo_b = (bf16*) alloc((size_t)Lc * DI * HID * 2);
    bf16*  wf_b = (bf16*) alloc((size_t)HID * DIMc * 2);

    auto cvt = [&](const float* s, bf16* d, size_t n) {
        size_t blocks = (n + 255) / 256;
        if (blocks > 4096) blocks = 4096;
        to_bf16_kernel<<<(int)blocks, 256, 0, stream>>>(s, d, n);
    };
    cvt(Wp,  wp_b,  (size_t)DIMc * HID);
    cvt(Whg, whg_b, (size_t)Lc * HID * 2 * DI);
    cvt(Wo,  wo_b,  (size_t)Lc * DI * HID);
    cvt(Wf,  wf_b,  (size_t)HID * DIMc);
    cvt(x,   ab,    (size_t)M * DIMc);

    auto gemm = [&](const bf16* A, const bf16* Wm, const float* bias,
                    float* outp, int N, int K) {
        dim3 grid(N / 128, M / 128);
        gemm_bf16_wmma_kernel<false><<<grid, 256, 0, stream>>>(A, Wm, bias, nullptr,
                                                               outp, N, K);
    };
    auto gemm_res = [&](const bf16* A, const bf16* Wm, const float* bias,
                        const float* res, float* outp, int N, int K) {
        dim3 grid(N / 128, M / 128);
        gemm_bf16_wmma_kernel<true><<<grid, 256, 0, stream>>>(A, Wm, bias, res,
                                                              outp, N, K);
    };

    // h = x @ Wp + bp
    gemm(ab, wp_b, bp, h, HID, DIMc);

    for (int i = 0; i < Lc; ++i) {
        // xn = rmsnorm(h, g1[i]) -> bf16
        rmsnorm_kernel<<<M, 256, 0, stream>>>(h, g1 + (size_t)i * HID, ab, HID, 0);
        // hg = xn @ Whg[i] + bhg[i]
        gemm(ab, whg_b + (size_t)i * HID * 2 * DI, bhg + (size_t)i * 2 * DI,
             hg, 2 * DI, HID);
        // log-space minGRU scan -> bf16 activations
        scan_kernel<<<(Bc * DI + 255) / 256, 256, 0, stream>>>(hg, ab, T, 2 * DI, DI,
                                                               Bc * DI);
        // h = scan @ Wo[i] + bo[i] + h
        gemm_res(ab, wo_b + (size_t)i * DI * HID, bo + (size_t)i * HID, h, h, HID, DI);
        // h = rmsnorm(h, g2[i]) + h  (in place)
        rmsnorm_kernel<<<M, 256, 0, stream>>>(h, g2 + (size_t)i * HID, h, HID, 1);
    }

    // out = rmsnorm(h, gf) @ Wf + bf
    rmsnorm_kernel<<<M, 256, 0, stream>>>(h, gf, ab, HID, 0);
    gemm(ab, wf_b, bfv, (float*)d_out, DIMc, HID);
}
